// GraphNN_43061342109882
// MI455X (gfx1250) — matrix operand order repaired and checked by reading, not compiled, not run on hardware
//
#include <hip/hip_runtime.h>
#include <hip/hip_bf16.h>

// ---------------------------------------------------------------------------
// GATv2 x2 on gfx1250: bf16 WMMA for node-feature GEMMs, scalar wave32 edge
// stage with segment softmax via float-key atomicMax + atomicAdd.
// ---------------------------------------------------------------------------

#define NN   50000
#define EE   800000
#define NDIM 128
#define EDIM 16
#define HH1  256
#define HH2  32
#define ETOT (EE + NN)          // edges + self loops
#define NEG_SLOPE 0.2f

typedef __attribute__((ext_vector_type(16))) __bf16 v16bf;
typedef __attribute__((ext_vector_type(8)))  float  v8f;

// ---- monotonic float <-> uint key for atomicMax-based segment max ----------
__device__ __forceinline__ unsigned fkey(float f) {
    unsigned u = __float_as_uint(f);
    return (u & 0x80000000u) ? ~u : (u | 0x80000000u);
}
__device__ __forceinline__ float funkey(unsigned u) {
    return (u & 0x80000000u) ? __uint_as_float(u & 0x7fffffffu)
                             : __uint_as_float(~u);
}

// ---------------------------------------------------------------------------
// small utility kernels
// ---------------------------------------------------------------------------
__global__ void fill_f32_kernel(float* p, int n, float v) {
    int i = blockIdx.x * blockDim.x + threadIdx.x;
    if (i < n) p[i] = v;
}
__global__ void fill_u32_kernel(unsigned* p, int n, unsigned v) {
    int i = blockIdx.x * blockDim.x + threadIdx.x;
    if (i < n) p[i] = v;
}
__global__ void f32_to_bf16_kernel(const float* __restrict__ s, __bf16* __restrict__ d, int n) {
    int i = blockIdx.x * blockDim.x + threadIdx.x;
    if (i < n) d[i] = (__bf16)s[i];
}
// h[n][c] = bias[c]
__global__ void init_bias_kernel(float* __restrict__ h, const float* __restrict__ bias,
                                 int C, int total) {
    int i = blockIdx.x * blockDim.x + threadIdx.x;
    if (i < total) h[i] = bias[i % C];
}

// ---------------------------------------------------------------------------
// self-loop attr: deg[d] += 1 ; loop_sum[d][j] += edge_attr[e][j]
// ---------------------------------------------------------------------------
__global__ void deg_loopsum_kernel(const int* __restrict__ dst_idx,
                                   const float* __restrict__ edge_attr,
                                   float* __restrict__ deg,
                                   float* __restrict__ loop_sum) {
    int t = blockIdx.x * blockDim.x + threadIdx.x;
    if (t >= EE * EDIM) return;
    int e = t >> 4, j = t & 15;
    int d = dst_idx[e];
    if (j == 0) atomicAdd(&deg[d], 1.0f);
    atomicAdd(&loop_sum[(size_t)d * EDIM + j], edge_attr[(size_t)e * EDIM + j]);
}
__global__ void loop_div_kernel(float* __restrict__ loop_attr, const float* __restrict__ deg) {
    int t = blockIdx.x * blockDim.x + threadIdx.x;
    if (t >= NN * EDIM) return;
    float dg = deg[t >> 4];
    loop_attr[t] = loop_attr[t] / fmaxf(dg, 1.0f);
}

// ---------------------------------------------------------------------------
// Y[nrows,Cout] = X[nrows,K] @ W[Cout,K]^T + bias    (bf16 WMMA, f32 accum)
// one wave per 16x16 output tile; K multiple of 32; nrows multiple of 16.
// ---------------------------------------------------------------------------
__global__ void linear_wmma_kernel(const float* __restrict__ X,
                                   const __bf16* __restrict__ Wbf,
                                   const float* __restrict__ bias,
                                   float* __restrict__ Y,
                                   int nrows, int K, int Cout) {
    const int lane = threadIdx.x & 31;
    const int warp = threadIdx.x >> 5;
    const int mt   = blockIdx.x * (blockDim.x >> 5) + warp;
    const int nt   = blockIdx.y;
    const int MT   = nrows >> 4;
    if (mt >= MT) return;                       // wave-uniform: EXEC all-ones below

    const int m  = lane & 15;                   // A row / D column / B column
    const int h4 = lane >> 4;                   // half-wave select
    const float*  xrow = X   + (size_t)(mt * 16 + m) * K;
    const __bf16* wrow = Wbf + (size_t)(nt * 16 + m) * K + h4 * 16;

    v8f c = {};
    for (int k0 = 0; k0 < K; k0 += 32) {
        v16bf a, b;
#pragma unroll
        for (int e = 0; e < 8; ++e)             // contiguous A layout: k = 16*h4 + e
            a[e] = (__bf16)xrow[k0 + 16 * h4 + e];
#pragma unroll
        for (int e = 8; e < 16; ++e)
            a[e] = (__bf16)xrow[k0 + 16 * h4 + e];
        b = *(const v16bf*)(wrow + k0);         // 16 contiguous bf16 (32B aligned)
        c = __builtin_amdgcn_wmma_f32_16x16x32_bf16(false, a, false, b,
                                                    (short)0, c, false, false);
    }
    const float bn = bias[nt * 16 + m];
#pragma unroll
    for (int r = 0; r < 8; ++r)                 // D: M = r + 8*h4, N = lane&15
        Y[(size_t)(mt * 16 + r + 8 * h4) * Cout + nt * 16 + m] = c[r] + bn;
}

// ---------------------------------------------------------------------------
// pass 1: logit[e] = att . leaky_relu(xl[src] + xr[dst] + ea@We^T); seg-max
// one wave per edge
// ---------------------------------------------------------------------------
template <int C>
__global__ void edge_logit_kernel(const float* __restrict__ xl, const float* __restrict__ xr,
                                  const int* __restrict__ src_idx, const int* __restrict__ dst_idx,
                                  const float* __restrict__ edge_attr,
                                  const float* __restrict__ loop_attr,
                                  const __bf16* __restrict__ We, const float* __restrict__ att,
                                  float* __restrict__ logit, unsigned* __restrict__ lmax_u) {
    const int e    = blockIdx.x * (blockDim.x >> 5) + (threadIdx.x >> 5);
    const int lane = threadIdx.x & 31;
    if (e >= ETOT) return;

    int s, d; const float* ea;
    if (e < EE) { s = src_idx[e]; d = dst_idx[e]; ea = edge_attr + (size_t)e * EDIM; }
    else        { s = d = e - EE;                 ea = loop_attr + (size_t)s * EDIM; }

    float eav[EDIM];
#pragma unroll
    for (int j = 0; j < EDIM; ++j) eav[j] = ea[j];   // broadcast reads, L2-served

    float acc = 0.0f;
    for (int c = lane; c < C; c += 32) {
        const __bf16* wr = We + c * EDIM;
        float ew = 0.0f;
#pragma unroll
        for (int j = 0; j < EDIM; ++j) ew = fmaf(eav[j], (float)wr[j], ew);
        float sum = xl[(size_t)s * C + c] + xr[(size_t)d * C + c] + ew;
        sum = sum > 0.0f ? sum : NEG_SLOPE * sum;
        acc = fmaf(sum, att[c], acc);
    }
#pragma unroll
    for (int off = 16; off > 0; off >>= 1)           // wave32 reduction
        acc += __shfl_xor(acc, off, 32);
    if (lane == 0) {
        logit[e] = acc;
        atomicMax(&lmax_u[d], fkey(acc));
    }
}

// pass 2: alpha = exp(logit - lmax[dst]); denom[dst] += alpha
__global__ void edge_alpha_kernel(const int* __restrict__ dst_idx,
                                  const float* __restrict__ logit,
                                  const unsigned* __restrict__ lmax_u,
                                  float* __restrict__ alpha, float* __restrict__ denom) {
    int e = blockIdx.x * blockDim.x + threadIdx.x;
    if (e >= ETOT) return;
    int d = (e < EE) ? dst_idx[e] : (e - EE);
    float a = __expf(logit[e] - funkey(lmax_u[d]));
    alpha[e] = a;
    atomicAdd(&denom[d], a);
}

// pass 3: h[dst] += xl[src] * alpha/denom[dst]   (h pre-initialized to bias)
template <int C>
__global__ void edge_aggr_kernel(const float* __restrict__ xl,
                                 const int* __restrict__ src_idx, const int* __restrict__ dst_idx,
                                 const float* __restrict__ alpha, const float* __restrict__ denom,
                                 float* __restrict__ h) {
    const int e    = blockIdx.x * (blockDim.x >> 5) + (threadIdx.x >> 5);
    const int lane = threadIdx.x & 31;
    if (e >= ETOT) return;
    int s, d;
    if (e < EE) { s = src_idx[e]; d = dst_idx[e]; }
    else        { s = d = e - EE; }
    const float w = alpha[e] / denom[d];
    for (int c = lane; c < C; c += 32)
        atomicAdd(&h[(size_t)d * C + c], xl[(size_t)s * C + c] * w);
}

// ---------------------------------------------------------------------------
// out[c] = mean over nodes of h2[n][c]; LDS ds_add_f32 partials per block
// ---------------------------------------------------------------------------
__global__ void mean_kernel(const float* __restrict__ h2, float* __restrict__ out) {
    __shared__ float s[HH2];
    const int t = threadIdx.x;
    if (t < HH2) s[t] = 0.0f;
    __syncthreads();
    const int rows = blockDim.x >> 5;
    const int row  = blockIdx.x * rows + (t >> 5);
    const int c    = t & 31;
    if (row < NN) atomicAdd(&s[c], h2[(size_t)row * HH2 + c]);
    __syncthreads();
    if (t < HH2) atomicAdd(&out[c], s[c] * (1.0f / (float)NN));
}

// ---------------------------------------------------------------------------
// host launcher
// ---------------------------------------------------------------------------
extern "C" void kernel_launch(void* const* d_in, const int* in_sizes, int n_in,
                              void* d_out, int out_size, void* d_ws, size_t ws_size,
                              hipStream_t stream) {
    (void)in_sizes; (void)n_in; (void)out_size; (void)ws_size;

    const float* x    = (const float*)d_in[0];
    const int*   ei   = (const int*)  d_in[1];      // [2, E]
    const float* ea   = (const float*)d_in[2];
    const float* Wl1  = (const float*)d_in[3];  const float* bl1   = (const float*)d_in[4];
    const float* Wr1  = (const float*)d_in[5];  const float* br1   = (const float*)d_in[6];
    const float* We1  = (const float*)d_in[7];  const float* att1  = (const float*)d_in[8];
    const float* bias1= (const float*)d_in[9];
    const float* Wl2  = (const float*)d_in[10]; const float* bl2   = (const float*)d_in[11];
    const float* Wr2  = (const float*)d_in[12]; const float* br2   = (const float*)d_in[13];
    const float* We2  = (const float*)d_in[14]; const float* att2  = (const float*)d_in[15];
    const float* bias2= (const float*)d_in[16];
    float* out = (float*)d_out;

    const int* src_idx = ei;
    const int* dst_idx = ei + EE;

    // ---- 256B-aligned bump allocator over d_ws ----
    size_t cur = 0;
    auto alloc = [&](size_t bytes) -> char* {
        char* p = (char*)d_ws + cur;
        cur = (cur + bytes + 255) & ~(size_t)255;
        return p;
    };
    __bf16* wl1b = (__bf16*)alloc((size_t)HH1 * NDIM * 2);
    __bf16* wr1b = (__bf16*)alloc((size_t)HH1 * NDIM * 2);
    __bf16* we1b = (__bf16*)alloc((size_t)HH1 * EDIM * 2);
    __bf16* wl2b = (__bf16*)alloc((size_t)HH2 * HH1 * 2);
    __bf16* wr2b = (__bf16*)alloc((size_t)HH2 * HH1 * 2);
    __bf16* we2b = (__bf16*)alloc((size_t)HH2 * EDIM * 2);
    float* deg    = (float*)alloc((size_t)NN * 4);
    float* loopat = (float*)alloc((size_t)NN * EDIM * 4);
    float* xl1    = (float*)alloc((size_t)NN * HH1 * 4);
    float* xr1    = (float*)alloc((size_t)NN * HH1 * 4);
    float* h1     = (float*)alloc((size_t)NN * HH1 * 4);
    float* xl2    = (float*)alloc((size_t)NN * HH2 * 4);
    float* xr2    = (float*)alloc((size_t)NN * HH2 * 4);
    float* h2     = (float*)alloc((size_t)NN * HH2 * 4);
    float* logit  = (float*)alloc((size_t)ETOT * 4);
    float* alpha  = (float*)alloc((size_t)ETOT * 4);
    unsigned* lmaxu = (unsigned*)alloc((size_t)NN * 4);
    float* denom  = (float*)alloc((size_t)NN * 4);

    const int B = 256;
    auto g1 = [&](int n) { return dim3((n + B - 1) / B); };

    // ---- weights -> bf16 ----
    f32_to_bf16_kernel<<<g1(HH1*NDIM), B, 0, stream>>>(Wl1, wl1b, HH1*NDIM);
    f32_to_bf16_kernel<<<g1(HH1*NDIM), B, 0, stream>>>(Wr1, wr1b, HH1*NDIM);
    f32_to_bf16_kernel<<<g1(HH1*EDIM), B, 0, stream>>>(We1, we1b, HH1*EDIM);
    f32_to_bf16_kernel<<<g1(HH2*HH1),  B, 0, stream>>>(Wl2, wl2b, HH2*HH1);
    f32_to_bf16_kernel<<<g1(HH2*HH1),  B, 0, stream>>>(Wr2, wr2b, HH2*HH1);
    f32_to_bf16_kernel<<<g1(HH2*EDIM), B, 0, stream>>>(We2, we2b, HH2*EDIM);

    // ---- self-loop attributes ----
    fill_f32_kernel<<<g1(NN), B, 0, stream>>>(deg, NN, 0.0f);
    fill_f32_kernel<<<g1(NN*EDIM), B, 0, stream>>>(loopat, NN*EDIM, 0.0f);
    deg_loopsum_kernel<<<g1(EE*EDIM), B, 0, stream>>>(dst_idx, ea, deg, loopat);
    loop_div_kernel<<<g1(NN*EDIM), B, 0, stream>>>(loopat, deg);

    const int MT = NN / 16;                     // 3125 tiles, 8 waves/block
    const dim3 gemmBlk(256);
    // ---- layer 1 GEMMs (WMMA) ----
    linear_wmma_kernel<<<dim3((MT+7)/8, HH1/16), gemmBlk, 0, stream>>>(x, wl1b, bl1, xl1, NN, NDIM, HH1);
    linear_wmma_kernel<<<dim3((MT+7)/8, HH1/16), gemmBlk, 0, stream>>>(x, wr1b, br1, xr1, NN, NDIM, HH1);

    // ---- layer 1 attention ----
    fill_u32_kernel<<<g1(NN), B, 0, stream>>>(lmaxu, NN, 0u);
    fill_f32_kernel<<<g1(NN), B, 0, stream>>>(denom, NN, 0.0f);
    const dim3 edgeGrid((ETOT + 7) / 8);        // 8 waves / block, wave per edge
    edge_logit_kernel<HH1><<<edgeGrid, 256, 0, stream>>>(xl1, xr1, src_idx, dst_idx,
                                                         ea, loopat, we1b, att1, logit, lmaxu);
    edge_alpha_kernel<<<g1(ETOT), B, 0, stream>>>(dst_idx, logit, lmaxu, alpha, denom);
    init_bias_kernel<<<g1(NN*HH1), B, 0, stream>>>(h1, bias1, HH1, NN*HH1);
    edge_aggr_kernel<HH1><<<edgeGrid, 256, 0, stream>>>(xl1, src_idx, dst_idx, alpha, denom, h1);

    // ---- layer 2 GEMMs (WMMA) ----
    linear_wmma_kernel<<<dim3((MT+7)/8, HH2/16), gemmBlk, 0, stream>>>(h1, wl2b, bl2, xl2, NN, HH1, HH2);
    linear_wmma_kernel<<<dim3((MT+7)/8, HH2/16), gemmBlk, 0, stream>>>(h1, wr2b, br2, xr2, NN, HH1, HH2);

    // ---- layer 2 attention ----
    fill_u32_kernel<<<g1(NN), B, 0, stream>>>(lmaxu, NN, 0u);
    fill_f32_kernel<<<g1(NN), B, 0, stream>>>(denom, NN, 0.0f);
    edge_logit_kernel<HH2><<<edgeGrid, 256, 0, stream>>>(xl2, xr2, src_idx, dst_idx,
                                                         ea, loopat, we2b, att2, logit, lmaxu);
    edge_alpha_kernel<<<g1(ETOT), B, 0, stream>>>(dst_idx, logit, lmaxu, alpha, denom);
    init_bias_kernel<<<g1(NN*HH2), B, 0, stream>>>(h2, bias2, HH2, NN*HH2);
    edge_aggr_kernel<HH2><<<edgeGrid, 256, 0, stream>>>(xl2, src_idx, dst_idx, alpha, denom, h2);

    // ---- mean over nodes ----
    fill_f32_kernel<<<dim3(1), 32, 0, stream>>>(out, HH2, 0.0f);
    mean_kernel<<<dim3((NN + 7) / 8), 256, 0, stream>>>(h2, out);
}
